// TemporalAttnLayer0_2_perfCeil_79542794322671
// MI455X (gfx1250) — matrix operand order repaired and checked by reading, not compiled, not run on hardware
//
#include <hip/hip_runtime.h>
#include <hip/hip_bf16.h>

// ---------------------------------------------------------------------------
// Problem constants (match reference)
// ---------------------------------------------------------------------------
#define NUM_HEADS 8
#define DIM_NODE  512
#define DIM_EDGE  172
#define DIM_TIME  100
#define DIM_OUT   512

// padded-K (multiple of 32) for the f16 staging copies
#define KP_NODE 512
#define KP_EDGE 192
#define KP_TIME 128

typedef _Float16 v16h __attribute__((ext_vector_type(16)));
typedef _Float16 v8h  __attribute__((ext_vector_type(8)));
typedef float    v8f  __attribute__((ext_vector_type(8)));

// ---------------------------------------------------------------------------
// Elementwise f32 -> f16 with K-padding:  dst(M,Kp) = pad(src(M,K))
// ---------------------------------------------------------------------------
__global__ void convert_pad_a16(const float* __restrict__ src,
                                _Float16* __restrict__ dst,
                                int M, int K, int Kp)
{
    long i = ((long)blockIdx.x * blockDim.x + threadIdx.x) * 2;  // pair index
    long total = (long)M * Kp;
    if (i >= total) return;
    int row = (int)(i / Kp);
    int k   = (int)(i % Kp);
    const float* s = src + (long)row * K;
    _Float16 a = (k     < K) ? (_Float16)s[k]     : (_Float16)0.0f;
    _Float16 b = (k + 1 < K) ? (_Float16)s[k + 1] : (_Float16)0.0f;
    dst[i]     = a;
    dst[i + 1] = b;
}

// ---------------------------------------------------------------------------
// Weight transpose + convert + K-pad:  wt(N,Kp) = pad(w(K,N)^T)
// ---------------------------------------------------------------------------
__global__ void transpose_pad_w16(const float* __restrict__ w,   // (K, N)
                                  _Float16* __restrict__ wt,     // (N, Kp)
                                  int K, int N, int Kp)
{
    int n = blockIdx.x;
    for (int k = threadIdx.x; k < Kp; k += blockDim.x)
        wt[(long)n * Kp + k] = (k < K) ? (_Float16)w[(long)k * N + n]
                                       : (_Float16)0.0f;
}

// ---------------------------------------------------------------------------
// Tiny GEMV: qt[n] = time_dst_unique[0,:] @ wq_time_w[:,n] + wq_time_b[n]
// ---------------------------------------------------------------------------
__global__ void qt_kernel(const float* __restrict__ t,
                          const float* __restrict__ w,
                          const float* __restrict__ b,
                          float* __restrict__ qt)
{
    int n = blockIdx.x * blockDim.x + threadIdx.x;
    if (n >= DIM_OUT) return;
    float acc = b[n];
    #pragma unroll 4
    for (int k = 0; k < DIM_TIME; ++k)
        acc += t[k] * w[k * DIM_OUT + n];
    qt[n] = acc;
}

// ---------------------------------------------------------------------------
// WMMA GEMM on pre-converted f16:  Y[M,N] = gather(A16)[M,Kp] @ Wt[N,Kp]^T + bias
//   Block: 256 threads = 8 waves; block tile 64(M) x 256(N); K step 32;
//   LDS double-buffered. Wave -> 32x64 patch (wm = w&1, wn = w>>1),
//   2x4 v_wmma_f32_16x16x32_f16 per K-step (8 WMMA / 12 ds_load_b128).
// ---------------------------------------------------------------------------
#define BM 64
#define BN 256
#define BK 32

__device__ __forceinline__ void wmma_step(const _Float16* __restrict__ sA,
                                          const _Float16* __restrict__ sB,
                                          int wm, int wn, int lr, int lh,
                                          v8f (&acc)[2][4])
{
    v16h afrag[2], bfrag[4];
    #pragma unroll
    for (int mi = 0; mi < 2; ++mi) {
        const _Float16* p = &sA[(wm + mi * 16 + lr) * BK + lh * 8];
        v8h lo = *(const v8h*)(p);
        v8h hi = *(const v8h*)(p + 16);
        #pragma unroll
        for (int q = 0; q < 8; ++q) { afrag[mi][q] = lo[q]; afrag[mi][q + 8] = hi[q]; }
    }
    #pragma unroll
    for (int ni = 0; ni < 4; ++ni) {
        const _Float16* p = &sB[(wn + ni * 16 + lr) * BK + lh * 8];
        v8h lo = *(const v8h*)(p);
        v8h hi = *(const v8h*)(p + 16);
        #pragma unroll
        for (int q = 0; q < 8; ++q) { bfrag[ni][q] = lo[q]; bfrag[ni][q + 8] = hi[q]; }
    }
    #pragma unroll
    for (int mi = 0; mi < 2; ++mi)
        #pragma unroll
        for (int ni = 0; ni < 4; ++ni)
            acc[mi][ni] = __builtin_amdgcn_wmma_f32_16x16x32_f16(
                false, afrag[mi], false, bfrag[ni],
                (short)0, acc[mi][ni], false, false);
}

__global__ __launch_bounds__(256)
void gemm_f16_wmma(const _Float16* __restrict__ A16,    // (M, Kp) f16, padded
                   const int*      __restrict__ gather, // nullptr -> identity
                   int M, int Kp,
                   const _Float16* __restrict__ Wt,     // (N, Kp) f16, padded
                   const float*    __restrict__ bias,   // (N)
                   float*          __restrict__ Y,      // (M, N)
                   int N)
{
    __shared__ __align__(16) _Float16 sA[2][BM * BK];   // row-major, stride BK
    __shared__ __align__(16) _Float16 sB[2][BN * BK];   // column-major: [n*BK+k]

    const int tid  = threadIdx.x;
    const int wave = tid >> 5;
    const int lane = tid & 31;
    const int lh   = lane >> 4;
    const int lr   = lane & 15;

    const int mBlk = blockIdx.x * BM;
    const int nBlk = blockIdx.y * BN;
    const int wm   = (wave & 1) * 32;       // wave M offset (2 waves in M)
    const int wn   = (wave >> 1) * 64;      // wave N offset (4 waves in N)

    v8f acc[2][4];
    #pragma unroll
    for (int i = 0; i < 2; ++i)
        #pragma unroll
        for (int j = 0; j < 4; ++j)
            acc[i][j] = (v8f)0.0f;

    // A staging: thread -> (row = tid>>2, 8-f16 chunk = (tid&3)*8)     [16B]
    const int arow = tid >> 2;
    const int akc  = (tid & 3) * 8;
    // B staging: thread -> one column (tid), whole 32-K span = 4 uint4 [64B]
    const int bcol = tid;

    // M-tail: clamp row instead of zero-fill — OOB rows are masked at store.
    const int gRow   = mBlk + arow;
    const int cRow   = (gRow < M) ? gRow : (M - 1);
    const long srcRow = gather ? (long)gather[cRow] : (long)cRow;
    const uint4* aSrc = (const uint4*)(A16 + srcRow * (long)Kp + akc);      // +kb/8
    const uint4* bSrc = (const uint4*)(Wt + (long)(nBlk + bcol) * Kp);      // +kb/8

    // ---- prologue: stage tile 0 into buffer 0 ----
    {
        uint4 aReg = aSrc[0];
        uint4 b0 = bSrc[0], b1 = bSrc[1], b2 = bSrc[2], b3 = bSrc[3];
        *(uint4*)&sA[0][arow * BK + akc] = aReg;
        uint4* d = (uint4*)&sB[0][bcol * BK];
        d[0] = b0; d[1] = b1; d[2] = b2; d[3] = b3;
    }
    __syncthreads();

    const int nIter = Kp / BK;
    for (int it = 1; it < nIter; ++it) {
        const int koff = it * (BK / 8);      // uint4 units (8 f16 each)
        // issue next-tile global loads (latency hidden behind WMMAs)
        uint4 aReg = aSrc[koff];
        uint4 b0 = bSrc[koff + 0], b1 = bSrc[koff + 1];
        uint4 b2 = bSrc[koff + 2], b3 = bSrc[koff + 3];

        wmma_step(sA[(it - 1) & 1], sB[(it - 1) & 1], wm, wn, lr, lh, acc);

        *(uint4*)&sA[it & 1][arow * BK + akc] = aReg;
        uint4* d = (uint4*)&sB[it & 1][bcol * BK];
        d[0] = b0; d[1] = b1; d[2] = b2; d[3] = b3;
        __syncthreads();
    }
    wmma_step(sA[(nIter - 1) & 1], sB[(nIter - 1) & 1], wm, wn, lr, lh, acc);

    // ---- epilogue: VGPR i holds M = i (lanes 0-15) / i+8 (lanes 16-31) ----
    const bool fullM = (mBlk + BM <= M);
    #pragma unroll
    for (int mi = 0; mi < 2; ++mi) {
        #pragma unroll
        for (int ni = 0; ni < 4; ++ni) {
            const int col = nBlk + wn + ni * 16 + lr;
            const float bv = bias[col];
            if (fullM) {
                #pragma unroll
                for (int i = 0; i < 8; ++i) {
                    int row = mBlk + wm + mi * 16 + lh * 8 + i;
                    Y[(long)row * N + col] = acc[mi][ni][i] + bv;
                }
            } else {
                #pragma unroll
                for (int i = 0; i < 8; ++i) {
                    int row = mBlk + wm + mi * 16 + lh * 8 + i;
                    if (row < M) Y[(long)row * N + col] = acc[mi][ni][i] + bv;
                }
            }
        }
    }
}

// ---------------------------------------------------------------------------
// q_dst[d, :] += qt[time_dst_inverse[d], :]
// ---------------------------------------------------------------------------
__global__ void qadd_kernel(float* __restrict__ qdst,
                            const float* __restrict__ qt,
                            const int* __restrict__ tdi,
                            int numDst)
{
    int d = blockIdx.x;
    if (d >= numDst) return;
    int c = threadIdx.x * 4;
    const float4 add = *(const float4*)(qt + (long)tdi[d] * DIM_OUT + c);
    float4* p = (float4*)(qdst + (long)d * DIM_OUT + c);
    float4 v = *p;
    v.x += add.x; v.y += add.y; v.z += add.z; v.w += add.w;
    *p = v;
}

// ---------------------------------------------------------------------------
// Per-edge: gather Q/K/V rows, per-head dot + leaky ReLU, pass V through.
// One wave per edge; lane covers 16 contiguous cols; 4-lane shfl_xor per head.
// ---------------------------------------------------------------------------
__global__ __launch_bounds__(256)
void attn_v_kernel(const int* __restrict__ idx,
                   const int* __restrict__ nInv,
                   const int* __restrict__ eInv,
                   const int* __restrict__ tInv,
                   const float* __restrict__ qdst,
                   const float* __restrict__ kvNode,
                   const float* __restrict__ kvEdge,
                   const float* __restrict__ kvTime,
                   float* __restrict__ outAttn,
                   float* __restrict__ outV,
                   int E)
{
    const int lane = threadIdx.x & 31;
    const long e = (long)blockIdx.x * 8 + (threadIdx.x >> 5);
    if (e >= E) return;

    const float4* q  = (const float4*)(qdst   + (long)idx[e]  * 512);
    const float4* kn = (const float4*)(kvNode + (long)nInv[e] * 1024);
    const float4* ke = (const float4*)(kvEdge + (long)eInv[e] * 1024);
    const float4* kt = (const float4*)(kvTime + (long)tInv[e] * 1024);
    float4* vo = (float4*)(outV + e * 512);

    const int f0 = lane * 4;
    float partial = 0.0f;

    #pragma unroll
    for (int j = 0; j < 4; ++j) {
        float4 a = kn[f0 + j], b = ke[f0 + j], c = kt[f0 + j];
        float4 kz = make_float4(a.x + b.x + c.x, a.y + b.y + c.y,
                                a.z + b.z + c.z, a.w + b.w + c.w);
        float4 qv = q[f0 + j];
        partial += qv.x * kz.x + qv.y * kz.y + qv.z * kz.z + qv.w * kz.w;

        float4 av = kn[128 + f0 + j], bv = ke[128 + f0 + j], cv = kt[128 + f0 + j];
        vo[f0 + j] = make_float4(av.x + bv.x + cv.x, av.y + bv.y + cv.y,
                                 av.z + bv.z + cv.z, av.w + bv.w + cv.w);
    }

    partial += __shfl_xor(partial, 1, 32);
    partial += __shfl_xor(partial, 2, 32);
    if ((lane & 3) == 0) {
        float a = partial;
        outAttn[e * NUM_HEADS + (lane >> 2)] = (a >= 0.0f) ? a : 0.2f * a;
    }
}

// ---------------------------------------------------------------------------
// Launcher
// ---------------------------------------------------------------------------
extern "C" void kernel_launch(void* const* d_in, const int* in_sizes, int n_in,
                              void* d_out, int out_size, void* d_ws, size_t ws_size,
                              hipStream_t stream)
{
    const int*   idx        = (const int*)  d_in[0];
    const float* nodeData   = (const float*)d_in[1];
    const int*   nodeInv    = (const int*)  d_in[2];
    const int*   nodeDstInv = (const int*)  d_in[3];
    const float* efeat      = (const float*)d_in[4];
    const int*   efeatInv   = (const int*)  d_in[5];
    const float* timeU      = (const float*)d_in[6];
    const int*   timeInv    = (const int*)  d_in[7];
    const float* timeDstU   = (const float*)d_in[8];
    const int*   timeDstInv = (const int*)  d_in[9];
    const float* wqNodeW    = (const float*)d_in[10];
    const float* wqNodeB    = (const float*)d_in[11];
    const float* wqTimeW    = (const float*)d_in[12];
    const float* wqTimeB    = (const float*)d_in[13];
    const float* wkvNodeW   = (const float*)d_in[14];
    const float* wkvNodeB   = (const float*)d_in[15];
    const float* wkvEdgeW   = (const float*)d_in[16];
    const float* wkvEdgeB   = (const float*)d_in[17];
    const float* wkvTimeW   = (const float*)d_in[18];
    const float* wkvTimeB   = (const float*)d_in[19];

    const int E      = in_sizes[0];
    const int U_NODE = in_sizes[1] / DIM_NODE;
    const int NUMDST = in_sizes[3];
    const int U_EDGE = in_sizes[4] / DIM_EDGE;
    const int U_TIME = in_sizes[6] / DIM_TIME;
    const int NKV    = 2 * DIM_OUT;

    // ---- workspace layout ----
    char* p = (char*)d_ws;
    float* qdst   = (float*)p;  p += (size_t)NUMDST * DIM_OUT * 4;
    float* qt     = (float*)p;  p += (size_t)DIM_OUT * 4;
    float* kvNode = (float*)p;  p += (size_t)U_NODE * NKV * 4;
    float* kvEdge = (float*)p;  p += (size_t)U_EDGE * NKV * 4;
    float* kvTime = (float*)p;  p += (size_t)U_TIME * NKV * 4;
    _Float16* a16Node = (_Float16*)p;  p += (size_t)U_NODE * KP_NODE * 2;
    _Float16* a16Edge = (_Float16*)p;  p += (size_t)U_EDGE * KP_EDGE * 2;
    _Float16* a16Time = (_Float16*)p;  p += (size_t)U_TIME * KP_TIME * 2;
    _Float16* wtQNode  = (_Float16*)p; p += (size_t)DIM_OUT * KP_NODE * 2;
    _Float16* wtKVNode = (_Float16*)p; p += (size_t)NKV * KP_NODE * 2;
    _Float16* wtKVEdge = (_Float16*)p; p += (size_t)NKV * KP_EDGE * 2;
    _Float16* wtKVTime = (_Float16*)p; p += (size_t)NKV * KP_TIME * 2;

    float* outAttn = (float*)d_out;
    float* outV    = outAttn + (long)E * NUM_HEADS;

    // ---- 0) one-time f16 conversions (activations padded, weights transposed)
    {
        long pairs;
        pairs = (long)U_NODE * KP_NODE / 2;
        convert_pad_a16<<<(int)((pairs + 255) / 256), 256, 0, stream>>>(
            nodeData, a16Node, U_NODE, DIM_NODE, KP_NODE);
        pairs = (long)U_EDGE * KP_EDGE / 2;
        convert_pad_a16<<<(int)((pairs + 255) / 256), 256, 0, stream>>>(
            efeat, a16Edge, U_EDGE, DIM_EDGE, KP_EDGE);
        pairs = (long)U_TIME * KP_TIME / 2;
        convert_pad_a16<<<(int)((pairs + 255) / 256), 256, 0, stream>>>(
            timeU, a16Time, U_TIME, DIM_TIME, KP_TIME);

        transpose_pad_w16<<<DIM_OUT, 128, 0, stream>>>(wqNodeW,  wtQNode,  DIM_NODE, DIM_OUT, KP_NODE);
        transpose_pad_w16<<<NKV,     128, 0, stream>>>(wkvNodeW, wtKVNode, DIM_NODE, NKV, KP_NODE);
        transpose_pad_w16<<<NKV,     128, 0, stream>>>(wkvEdgeW, wtKVEdge, DIM_EDGE, NKV, KP_EDGE);
        transpose_pad_w16<<<NKV,     128, 0, stream>>>(wkvTimeW, wtKVTime, DIM_TIME, NKV, KP_TIME);
    }

    // ---- 1) qt = time_dst_unique @ wq_time_w + b
    qt_kernel<<<1, 512, 0, stream>>>(timeDstU, wqTimeW, wqTimeB, qt);

    // ---- 2) q_dst = nodeData[node_dst_inverse] @ wq_node_w + b (gathered)
    {
        dim3 grid((NUMDST + BM - 1) / BM, DIM_OUT / BN);
        gemm_f16_wmma<<<grid, 256, 0, stream>>>(a16Node, nodeDstInv,
                                                NUMDST, KP_NODE,
                                                wtQNode, wqNodeB, qdst, DIM_OUT);
    }
    // ---- 3) q_dst += qt[time_dst_inverse]
    qadd_kernel<<<NUMDST, 128, 0, stream>>>(qdst, qt, timeDstInv, NUMDST);

    // ---- 4..6) kv projections over unique rows
    {
        dim3 grid((U_NODE + BM - 1) / BM, NKV / BN);
        gemm_f16_wmma<<<grid, 256, 0, stream>>>(a16Node, nullptr, U_NODE, KP_NODE,
                                                wtKVNode, wkvNodeB, kvNode, NKV);
    }
    {
        dim3 grid((U_EDGE + BM - 1) / BM, NKV / BN);
        gemm_f16_wmma<<<grid, 256, 0, stream>>>(a16Edge, nullptr, U_EDGE, KP_EDGE,
                                                wtKVEdge, wkvEdgeB, kvEdge, NKV);
    }
    {
        dim3 grid((U_TIME + BM - 1) / BM, NKV / BN);
        gemm_f16_wmma<<<grid, 256, 0, stream>>>(a16Time, nullptr, U_TIME, KP_TIME,
                                                wtKVTime, wkvTimeB, kvTime, NKV);
    }

    // ---- 7) per-edge gather + per-head dot + leaky ReLU + V passthrough
    {
        int blocks = (E + 7) / 8;
        attn_v_kernel<<<blocks, 256, 0, stream>>>(idx, nodeInv, efeatInv, timeInv,
                                                  qdst, kvNode, kvEdge, kvTime,
                                                  outAttn, outV, E);
    }
}
